// MaskedMultiHeadSelfAttention_21895743275108
// MI455X (gfx1250) — matrix operand order repaired and checked by reading, not compiled, not run on hardware
//
#include <hip/hip_runtime.h>

// ---------------------------------------------------------------------------
// Masked multi-head self-attention with edge bias, CDNA5 (gfx1250) WMMA path.
// B=16, N=512, F=256, H=8, D=32, FO=32.  bf16 WMMA (16x16x32) with f32 acc.
// Round 2: cooperative async global->LDS staging (ASYNCcnt) for shared tiles:
//   - attention: K/V tiles double-buffered in LDS (8 waves shared them before)
//   - projections: x-tile staged once per block
// ---------------------------------------------------------------------------

#define B_  16
#define N_  512
#define F_  256
#define H_  8
#define D_  32
#define FO_ 32
#define C_  256   // H_*D_ == H_*FO_
#define BN_ (B_ * N_)        // 8192 rows
#define INV_SQRT_D 0.17677669529663687f

typedef __attribute__((ext_vector_type(16))) __bf16 v16bf;
typedef __attribute__((ext_vector_type(8)))  float  v8f;

union FragAB {
  v16bf v;
  uint4 u[2];
};

static __device__ __forceinline__ float red_max16(float v) {
  v = fmaxf(v, __shfl_xor(v, 8, 32));
  v = fmaxf(v, __shfl_xor(v, 4, 32));
  v = fmaxf(v, __shfl_xor(v, 2, 32));
  v = fmaxf(v, __shfl_xor(v, 1, 32));
  return v;
}
static __device__ __forceinline__ float red_sum16(float v) {
  v += __shfl_xor(v, 8, 32);
  v += __shfl_xor(v, 4, 32);
  v += __shfl_xor(v, 2, 32);
  v += __shfl_xor(v, 1, 32);
  return v;
}
static __device__ __forceinline__ float red_sum32(float v) {
  v += __shfl_xor(v, 16, 32);
  return red_sum16(v);
}

static __device__ __forceinline__ v8f wmma_bf16(const v16bf& a, const v16bf& b, const v8f& c) {
  return __builtin_amdgcn_wmma_f32_16x16x32_bf16(false, a, false, b, (short)0, c, false, false);
}

// Async 16-byte global -> LDS copy (GLOBAL_LOAD_ASYNC_TO_LDS_B128, ASYNCcnt).
// lds_off: LDS byte offset (low 32 bits of generic shared address, per the
// ISA aperture mapping).  gaddr: 64-bit global address.
static __device__ __forceinline__ void async_copy_b128(uint32_t lds_off, uint64_t gaddr) {
  asm volatile("global_load_async_to_lds_b128 %0, %1, off"
               :: "v"(lds_off), "v"(gaddr) : "memory");
}
static __device__ __forceinline__ void wait_asynccnt0() {
  asm volatile("s_wait_asynccnt 0x0" ::: "memory");
}
static __device__ __forceinline__ uint32_t lds_offset_of(const void* p) {
  return (uint32_t)(uintptr_t)p;   // generic shared addr: low 32 bits = LDS offset
}

// ---------------------------------------------------------------------------
// Kernel 1: transpose 4 weight matrices [F,256] f32 -> [256,F] bf16.
// w_q gets 1/sqrt(D) folded in.  wT_all = 4 matrices of 256*256 bf16.
// ---------------------------------------------------------------------------
__global__ __launch_bounds__(256)
void prep_weights_kernel(const float* __restrict__ wq, const float* __restrict__ wk,
                         const float* __restrict__ wv, const float* __restrict__ wg,
                         __bf16* __restrict__ wT_all) {
  int idx = blockIdx.x * 256 + threadIdx.x;   // 0 .. 4*65536-1
  int m   = idx >> 16;
  int rem = idx & 65535;
  int c   = rem >> 8;
  int f   = rem & 255;
  const float* W = (m == 0) ? wq : (m == 1) ? wk : (m == 2) ? wv : wg;
  float scale = (m == 0) ? INV_SQRT_D : 1.0f;
  wT_all[(size_t)m * 65536 + (size_t)c * 256 + f] = (__bf16)(W[(size_t)f * 256 + c] * scale);
}

// ---------------------------------------------------------------------------
// Kernel 2: parameter-free LayerNorm over F, multiply by mask, emit bf16 x.
// One wave (32 lanes) per row; each lane owns 8 contiguous floats.
// ---------------------------------------------------------------------------
__global__ __launch_bounds__(256)
void ln_kernel(const float* __restrict__ feats, const int* __restrict__ mask,
               __bf16* __restrict__ x_bf) {
  int row  = blockIdx.x * 8 + (threadIdx.x >> 5);
  int lane = threadIdx.x & 31;
  const float* fr = feats + (size_t)row * F_ + lane * 8;
  float4 a = *(const float4*)(fr);
  float4 b = *(const float4*)(fr + 4);
  float s  = a.x + a.y + a.z + a.w + b.x + b.y + b.z + b.w;
  float sq = a.x*a.x + a.y*a.y + a.z*a.z + a.w*a.w
           + b.x*b.x + b.y*b.y + b.z*b.z + b.w*b.w;
  s  = red_sum32(s);
  sq = red_sum32(sq);
  float mu  = s * (1.0f / F_);
  float var = sq * (1.0f / F_) - mu * mu;
  float rs  = rsqrtf(var + 1e-5f);
  float mf  = (mask[row] != 0) ? 1.0f : 0.0f;
  rs *= mf;
  __bf16* xr = x_bf + (size_t)row * F_ + lane * 8;
  xr[0] = (__bf16)((a.x - mu) * rs);
  xr[1] = (__bf16)((a.y - mu) * rs);
  xr[2] = (__bf16)((a.z - mu) * rs);
  xr[3] = (__bf16)((a.w - mu) * rs);
  xr[4] = (__bf16)((b.x - mu) * rs);
  xr[5] = (__bf16)((b.y - mu) * rs);
  xr[6] = (__bf16)((b.z - mu) * rs);
  xr[7] = (__bf16)((b.w - mu) * rs);
}

// ---------------------------------------------------------------------------
// Kernel 3: projections q,k,v,gate = x @ W  via bf16 WMMA 16x16x32, f32 acc.
// grid = (BN/16 mtiles, 4 matrices), 256 threads = 8 waves.
// The 16-row x tile (8KB, shared by all 8 waves) is async-staged into LDS.
// Layouts produced:
//   q_bf, k_bf : [B][H][N][D]  bf16   (q pre-scaled by 1/sqrt(D))
//   vT_bf      : [B][H][FO][N] bf16   (transposed for contiguous attention B-frags)
//   gate_f     : [BN][C] f32 (+ b_g)
// ---------------------------------------------------------------------------
static __device__ __forceinline__
void store_proj_tile(int mat, int mtile, int nt, int l16, int hi, const v8f& acc,
                     const float* __restrict__ b_g,
                     __bf16* __restrict__ q_bf, __bf16* __restrict__ k_bf,
                     __bf16* __restrict__ vT_bf, float* __restrict__ gate_f) {
  int col = nt * 16 + l16;
#pragma unroll
  for (int r = 0; r < 8; ++r) {
    int bn = mtile * 16 + hi * 8 + r;
    float val = acc[r];
    int b = bn >> 9, n = bn & 511;
    if (mat == 0) {
      int h = col >> 5, d = col & 31;
      q_bf[(((size_t)(b * H_ + h)) * N_ + n) * D_ + d] = (__bf16)val;
    } else if (mat == 1) {
      int h = col >> 5, d = col & 31;
      k_bf[(((size_t)(b * H_ + h)) * N_ + n) * D_ + d] = (__bf16)val;
    } else if (mat == 2) {
      int h = col >> 5, fo = col & 31;
      vT_bf[(((size_t)(b * H_ + h)) * FO_ + fo) * N_ + n] = (__bf16)val;
    } else {
      gate_f[(size_t)bn * C_ + col] = val + b_g[col];
    }
  }
}

__global__ __launch_bounds__(256)
void proj_kernel(const __bf16* __restrict__ x_bf, const __bf16* __restrict__ wT_all,
                 const float* __restrict__ b_g,
                 __bf16* __restrict__ q_bf, __bf16* __restrict__ k_bf,
                 __bf16* __restrict__ vT_bf, float* __restrict__ gate_f) {
  __shared__ __align__(16) __bf16 xs[16][F_];   // 8 KB x-tile, shared by all waves

  int mat   = blockIdx.y;
  int mtile = blockIdx.x;                  // 0..511
  int w     = threadIdx.x >> 5;
  int lane  = threadIdx.x & 31;
  int hi    = lane >> 4;                   // half-wave id (K split)
  int l16   = lane & 15;
  int nt0 = w, nt1 = w + 8;

  // async stage the 16x256 bf16 x tile (contiguous 8KB): 2 b128 per thread
  {
    const __bf16* xsrc = x_bf + (size_t)mtile * 16 * F_;
    uint32_t xl = lds_offset_of(&xs[0][0]);
    uint32_t o  = (uint32_t)threadIdx.x * 32u;
    uint64_t g  = (uint64_t)(uintptr_t)xsrc + o;
    async_copy_b128(xl + o,       g);
    async_copy_b128(xl + o + 16u, g + 16u);
    wait_asynccnt0();
    __syncthreads();
  }

  const __bf16* wT = wT_all + (size_t)mat * 65536;
  v8f acc0 = {}, acc1 = {};

#pragma unroll
  for (int kk = 0; kk < 8; ++kk) {
    int kbase = kk * 32;
    FragAB a, b0, b1;
    // A frag from LDS: per lane row = l16, 16 contiguous K values (k = 16*hi + s)
    const __bf16* xp = &xs[l16][kbase + hi * 16];
    a.u[0] = *(const uint4*)(xp);
    a.u[1] = *(const uint4*)(xp + 8);
    // B frags: per lane col = nt*16+l16, 16 contiguous K values (col-major wT)
    const __bf16* wp0 = wT + (size_t)(nt0 * 16 + l16) * F_ + kbase + hi * 16;
    b0.u[0] = *(const uint4*)(wp0);
    b0.u[1] = *(const uint4*)(wp0 + 8);
    const __bf16* wp1 = wT + (size_t)(nt1 * 16 + l16) * F_ + kbase + hi * 16;
    b1.u[0] = *(const uint4*)(wp1);
    b1.u[1] = *(const uint4*)(wp1 + 8);
    acc0 = wmma_bf16(a.v, b0.v, acc0);
    acc1 = wmma_bf16(a.v, b1.v, acc1);
  }
  store_proj_tile(mat, mtile, nt0, l16, hi, acc0, b_g, q_bf, k_bf, vT_bf, gate_f);
  store_proj_tile(mat, mtile, nt1, l16, hi, acc1, b_g, q_bf, k_bf, vT_bf, gate_f);
}

// ---------------------------------------------------------------------------
// Kernel 4: flash-style attention. grid = B*H*(N/128) = 512 blocks,
// 256 threads = 8 waves; each wave owns one 16-query tile, streams 32 keys
// per step (two 16x16x32 WMMAs for logits, two for attn@V).
// K-tile (32x32 bf16, 2KB) and V-tile (32x32 bf16, 2KB) are shared by all
// 8 waves -> double-buffered async staging into LDS (one b128 per thread).
// Pipeline per step: wait_async(0) ; barrier ; issue stage(kk+1) ; consume(kk).
// ---------------------------------------------------------------------------
__global__ __launch_bounds__(256)
void MaskedMultiHeadSelfAttention_21895743275108_kernel(
    const __bf16* __restrict__ q_bf, const __bf16* __restrict__ k_bf,
    const __bf16* __restrict__ vT_bf, const float* __restrict__ gate_f,
    const float* __restrict__ edges, const int* __restrict__ mask,
    const float* __restrict__ w_edge, float* __restrict__ out) {
  __shared__ __align__(16) __bf16 ptile[8][16][32];   // prob tile per wave, 8KB
  __shared__ __align__(16) __bf16 ktile[2][32][32];   // [key][d]  double buf, 4KB
  __shared__ __align__(16) __bf16 vtile[2][32][32];   // [fo][key] double buf, 4KB

  int bid    = blockIdx.x;                 // 0..511
  int qchunk = bid & 3;
  int h      = (bid >> 2) & 7;
  int b      = bid >> 5;
  int tid    = threadIdx.x;
  int w      = tid >> 5;
  int lane   = tid & 31;
  int hi     = lane >> 4;
  int l16    = lane & 15;
  int q0     = (qchunk * 8 + w) * 16;      // first query row of this wave
  size_t bh  = (size_t)(b * H_ + h);

  const __bf16* ksrc = k_bf + bh * N_ * D_;      // k-tile kk: +kk*32*D_, 2KB contiguous
  const __bf16* vsrc = vT_bf + bh * FO_ * N_;    // rows stride N_
  uint32_t k_lds = lds_offset_of(&ktile[0][0][0]);
  uint32_t v_lds = lds_offset_of(&vtile[0][0][0]);

  // cooperative stage of K/V tile for step kk into buffer nb (one b128/thread)
  auto stage = [&](int kk, int nb) {
    if (tid < 128) {
      // K tile: 2048 contiguous bytes, chunk = tid
      uint64_t g = (uint64_t)(uintptr_t)(ksrc + (size_t)kk * 32 * D_) + (uint32_t)tid * 16u;
      async_copy_b128(k_lds + (uint32_t)nb * 2048u + (uint32_t)tid * 16u, g);
    } else {
      // V tile: 32 rows (fo) of 64B, row stride N_*2 bytes; 4 chunks per row
      int t  = tid - 128;
      int fo = t >> 2;
      int ce = (t & 3) * 8;                // element offset within row
      uint64_t g = (uint64_t)(uintptr_t)(vsrc + (size_t)fo * N_ + (size_t)kk * 32 + ce);
      async_copy_b128(v_lds + (uint32_t)nb * 2048u + (uint32_t)t * 16u, g);
    }
  };

  // Q A-fragment (persistent over the K loop)
  FragAB aq;
  {
    const __bf16* qp = q_bf + (bh * N_ + q0 + l16) * D_ + hi * 16;
    aq.u[0] = *(const uint4*)(qp);
    aq.u[1] = *(const uint4*)(qp + 8);
  }
  float we = w_edge[h];
  int mq[8];
#pragma unroll
  for (int r = 0; r < 8; ++r)
    mq[r] = mask[b * N_ + q0 + hi * 8 + r];

  float m_run[8], s_run[8];
#pragma unroll
  for (int r = 0; r < 8; ++r) { m_run[r] = -INFINITY; s_run[r] = 0.0f; }
  v8f acc0 = {}, acc1 = {};

  stage(0, 0);
  int cb = 0;
  for (int kk = 0; kk < 16; ++kk) {
    int kbase = kk * 32;
    wait_asynccnt0();        // this wave's chunks of stage(kk) landed
    __syncthreads();         // all waves' chunks landed; prev-buffer readers done
    if (kk + 1 < 16) stage(kk + 1, cb ^ 1);   // overlap DMA with compute

    // K^T B-fragments from LDS: per lane col=key, 16 contiguous d values
    FragAB bk0, bk1;
    const __bf16* kb0 = &ktile[cb][l16][hi * 16];
    bk0.u[0] = *(const uint4*)(kb0);
    bk0.u[1] = *(const uint4*)(kb0 + 8);
    const __bf16* kb1 = &ktile[cb][16 + l16][hi * 16];
    bk1.u[0] = *(const uint4*)(kb1);
    bk1.u[1] = *(const uint4*)(kb1 + 8);

    v8f c0 = {}, c1 = {};
    c0 = wmma_bf16(aq.v, bk0.v, c0);       // logits, keys kbase..kbase+15
    c1 = wmma_bf16(aq.v, bk1.v, c1);       // logits, keys kbase+16..+31

    int mk0 = mask[b * N_ + kbase + l16];
    int mk1 = mask[b * N_ + kbase + 16 + l16];

    // prefetch next edges tile into cache (global_prefetch_b8)
    if (kk + 1 < 16)
      __builtin_prefetch(edges + ((size_t)(b * N_ + q0 + hi * 8)) * N_ + kbase + 32 + l16, 0, 1);

#pragma unroll
    for (int r = 0; r < 8; ++r) {
      int qi = q0 + hi * 8 + r;
      const float* ep = edges + ((size_t)(b * N_ + qi)) * N_ + kbase + l16;
      float e0 = ep[0], e1 = ep[16];
      bool v0 = (mq[r] != 0) && (mk0 != 0);
      bool v1 = (mq[r] != 0) && (mk1 != 0);
      float ef0 = v0 ? e0 : 0.0f;
      float ef1 = v1 ? e1 : 0.0f;
      // edge_bias = -|ef*w| + log_sigmoid(8-ef) = -|ef*w| - log(1+exp(ef-8))
      float bias0 = -fabsf(ef0 * we) - __logf(1.0f + __expf(ef0 - 8.0f));
      float bias1 = -fabsf(ef1 * we) - __logf(1.0f + __expf(ef1 - 8.0f));
      float l0 = c0[r] + bias0 + (v0 ? 0.0f : -1e20f);
      float l1 = c1[r] + bias1 + (v1 ? 0.0f : -1e20f);

      // online softmax across the 16 lanes of this half-wave (row-wise)
      float tm   = red_max16(fmaxf(l0, l1));
      float mnew = fmaxf(m_run[r], tm);
      float sc   = __expf(m_run[r] - mnew);
      float p0   = __expf(l0 - mnew);
      float p1   = __expf(l1 - mnew);
      float ts   = red_sum16(p0 + p1);
      s_run[r]   = s_run[r] * sc + ts;
      m_run[r]   = mnew;
      acc0[r] *= sc;
      acc1[r] *= sc;
      // C-layout -> LDS (row-major 16x32 bf16 prob tile)
      ptile[w][hi * 8 + r][l16]      = (__bf16)p0;
      ptile[w][hi * 8 + r][16 + l16] = (__bf16)p1;
    }

    __builtin_amdgcn_wave_barrier();       // intra-wave LDS exchange ordering
    FragAB ap;                             // prob A-fragment (16x32 bf16)
    {
      const __bf16* pp = &ptile[w][l16][hi * 16];
      ap.u[0] = *(const uint4*)(pp);
      ap.u[1] = *(const uint4*)(pp + 8);
    }
    __builtin_amdgcn_wave_barrier();

    // V B-fragments from LDS: per lane col=fo, 16 contiguous k values
    FragAB bv0, bv1;
    const __bf16* vb0 = &vtile[cb][l16][hi * 16];
    bv0.u[0] = *(const uint4*)(vb0);
    bv0.u[1] = *(const uint4*)(vb0 + 8);
    const __bf16* vb1 = &vtile[cb][16 + l16][hi * 16];
    bv1.u[0] = *(const uint4*)(vb1);
    bv1.u[1] = *(const uint4*)(vb1 + 8);

    acc0 = wmma_bf16(ap.v, bv0.v, acc0);   // out cols fo 0..15
    acc1 = wmma_bf16(ap.v, bv1.v, acc1);   // out cols fo 16..31

    cb ^= 1;
  }

  // epilogue: normalize, gate with sigmoid, store f32 [b, n, h*FO + fo]
#pragma unroll
  for (int r = 0; r < 8; ++r) {
    int qi = q0 + hi * 8 + r;
    float inv = 1.0f / s_run[r];
    size_t obase = ((size_t)(b * N_ + qi)) * C_ + h * FO_;
    float g0 = gate_f[obase + l16];
    float g1 = gate_f[obase + 16 + l16];
    out[obase + l16]      = acc0[r] * inv * (1.0f / (1.0f + __expf(-g0)));
    out[obase + 16 + l16] = acc1[r] * inv * (1.0f / (1.0f + __expf(-g1)));
  }
}

// ---------------------------------------------------------------------------
// Host launcher
// ---------------------------------------------------------------------------
extern "C" void kernel_launch(void* const* d_in, const int* in_sizes, int n_in,
                              void* d_out, int out_size, void* d_ws, size_t ws_size,
                              hipStream_t stream) {
  const float* feats  = (const float*)d_in[0];
  const int*   mask   = (const int*)  d_in[1];
  const float* edges  = (const float*)d_in[2];
  const float* w_q    = (const float*)d_in[3];
  const float* w_k    = (const float*)d_in[4];
  const float* w_v    = (const float*)d_in[5];
  const float* w_g    = (const float*)d_in[6];
  const float* b_g    = (const float*)d_in[7];
  const float* w_edge = (const float*)d_in[8];
  float* out = (float*)d_out;

  // workspace carve-up (all sections 256B-aligned by construction; ~24.5 MB)
  char* ws = (char*)d_ws;
  size_t off = 0;
  __bf16* x_bf   = (__bf16*)(ws + off); off += (size_t)BN_ * F_ * 2;          // 4 MB
  __bf16* wT_all = (__bf16*)(ws + off); off += (size_t)4 * 65536 * 2;         // 0.5 MB
  __bf16* q_bf   = (__bf16*)(ws + off); off += (size_t)B_ * H_ * N_ * D_ * 2; // 4 MB
  __bf16* k_bf   = (__bf16*)(ws + off); off += (size_t)B_ * H_ * N_ * D_ * 2; // 4 MB
  __bf16* vT_bf  = (__bf16*)(ws + off); off += (size_t)B_ * H_ * FO_ * N_ * 2;// 4 MB
  float*  gate_f = (float*) (ws + off); off += (size_t)BN_ * C_ * 4;          // 8 MB
  (void)ws_size; (void)in_sizes; (void)n_in; (void)out_size;

  prep_weights_kernel<<<1024, 256, 0, stream>>>(w_q, w_k, w_v, w_g, wT_all);
  ln_kernel<<<1024, 256, 0, stream>>>(feats, mask, x_bf);
  proj_kernel<<<dim3(512, 4), 256, 0, stream>>>(x_bf, wT_all, b_g,
                                                q_bf, k_bf, vT_bf, gate_f);
  MaskedMultiHeadSelfAttention_21895743275108_kernel<<<512, 256, 0, stream>>>(
      q_bf, k_bf, vT_bf, gate_f, edges, mask, w_edge, out);
}
